// LSTMSeq2Seq_66099546685878
// MI455X (gfx1250) — matrix operand order repaired and checked by reading, not compiled, not run on hardware
//
#include <hip/hip_runtime.h>
#include <cstdint>
#include <cstddef>

// ---------------------------------------------------------------------------
// Types / fragments for V_WMMA_F32_16X16X32_BF16 (wave32)
// ---------------------------------------------------------------------------
typedef __attribute__((ext_vector_type(16))) __bf16          v16bf;
typedef __attribute__((ext_vector_type(8)))  float           v8f;
typedef __attribute__((ext_vector_type(8)))  unsigned short  u16x8;
typedef int v4i_t __attribute__((vector_size(16)));   // matches builtin prototype

union Frag { v16bf v; u16x8 h[2]; };

__device__ __forceinline__ float sigm(float x)      { return 1.0f / (1.0f + __expf(-x)); }
__device__ __forceinline__ float tanh_fast(float x) { return 2.0f / (1.0f + __expf(-2.0f * x)) - 1.0f; }

// Problem constants
constexpr int  cV = 2048, cE = 768, cH = 768, cL = 2, cB = 256, cT = 128;
constexpr int  cG4 = 4 * cH;                 // 3072
constexpr long cMTOK = (long)cB * cT;        // 32768

// K-slice staged in LDS per pipeline stage, padded LDS row stride.
// Stride 40 elems (80B): fragment row accesses map to 16 distinct bank groups.
constexpr int KS   = 32;
constexpr int SROW = 40;

// ---------------------------------------------------------------------------
// 16-byte global -> LDS staging: async DMA path on CDNA5, ds_store fallback.
// ---------------------------------------------------------------------------
#if defined(__has_builtin)
#if __has_builtin(__builtin_amdgcn_global_load_async_to_lds_b128) && \
    __has_builtin(__builtin_amdgcn_s_wait_asynccnt)
#define ASYNC_LDS 1
#endif
#endif

__device__ __forceinline__ void stage16(const __bf16* g, __bf16* l)
{
#ifdef ASYNC_LDS
  __builtin_amdgcn_global_load_async_to_lds_b128(
      (__attribute__((address_space(1))) v4i_t*)(v4i_t*)g,
      (__attribute__((address_space(3))) v4i_t*)(v4i_t*)l,
      0, 0);
#else
  *(uint4*)l = *(const uint4*)g;
#endif
}

#ifdef ASYNC_LDS
#define WAIT_ASYNC(n) __builtin_amdgcn_s_wait_asynccnt(n)
#else
#define WAIT_ASYNC(n) ((void)0)
#endif

// Fragment loads from LDS (row-major staged tile, padded stride SROW).
// A 16x32 frag: grp0 -> K {0..15}, grp1 -> K {16..31}
__device__ __forceinline__ v16bf frag_a_lds(const __bf16* rowp, int grp)
{
  Frag f;
  const __bf16* p = rowp + grp * 16;
  f.h[0] = *(const u16x8*)(p);
  f.h[1] = *(const u16x8*)(p + 8);
  return f.v;
}

// ---------------------------------------------------------------------------
// Block-tiled GEMM: C[m,n] = sum_k A[m,k]*B[n,k] + bias0[n] (+bias1[n])
// 128x128 C tile per 256-thread block; 8 waves each own 32(M) x 64(N);
// A/B staged in LDS in K=32 slices, double buffered via async-to-LDS.
// Requires M%128==0, N%128==0, K%KS==0.
// ---------------------------------------------------------------------------
__global__ void __launch_bounds__(256)
gemm_tile(const __bf16* __restrict__ A,  int lda,
          const __bf16* __restrict__ Bw, int ldb,
          const float*  __restrict__ bias0,
          const float*  __restrict__ bias1,
          float* __restrict__ C, int ldc,
          int M, int N, int K)
{
  __shared__ __bf16 As[2][128][SROW];
  __shared__ __bf16 Bs[2][128][SROW];

  const int tid  = threadIdx.x;
  const int lane = tid & 31;
  const int widx = tid >> 5;
  const int nb   = N >> 7;
  const int bm   = blockIdx.x / nb;
  const int bn   = blockIdx.x % nb;
  const int row  = lane & 15, grp = lane >> 4;
  const int m_off = (widx & 3) << 5;        // 0,32,64,96
  const int n_off = (widx >> 2) << 6;       // 0,64

  auto stage = [&](int buf, int k0) {
#pragma unroll
    for (int i = 0; i < 2; ++i) {
      const int idx = i * 256 + tid;        // 0..511
      const int r   = idx >> 2;             // 0..127
      const int sg  = idx & 3;              // 16B segment within K slice
      stage16(A  + (size_t)(bm * 128 + r) * lda + k0 + sg * 8, &As[buf][r][sg * 8]);
      stage16(Bw + (size_t)(bn * 128 + r) * ldb + k0 + sg * 8, &Bs[buf][r][sg * 8]);
    }
  };

  v8f acc[2][4] = {};
  const int nst = K / KS;
  stage(0, 0);
  for (int s = 0; s < nst; ++s) {
    const int buf = s & 1;
    if (s + 1 < nst) { stage(buf ^ 1, (s + 1) * KS); WAIT_ASYNC(4); }
    else             { WAIT_ASYNC(0); }
    __syncthreads();

    Frag fb[4];
#pragma unroll
    for (int j = 0; j < 4; ++j) {
      const __bf16* p = &Bs[buf][n_off + (j << 4) + row][grp * 16];
      fb[j].h[0] = *(const u16x8*)(p);
      fb[j].h[1] = *(const u16x8*)(p + 8);
    }
#pragma unroll
    for (int mi = 0; mi < 2; ++mi) {
      v16bf fa = frag_a_lds(&As[buf][m_off + (mi << 4) + row][0], grp);
#pragma unroll
      for (int j = 0; j < 4; ++j)
        acc[mi][j] = __builtin_amdgcn_wmma_f32_16x16x32_bf16(
            false, fa, false, fb[j].v, (short)0, acc[mi][j], false, false);
    }
    __syncthreads();
  }

  const int m0 = bm * 128 + m_off + (grp << 3);
#pragma unroll
  for (int mi = 0; mi < 2; ++mi)
#pragma unroll
    for (int j = 0; j < 4; ++j) {
      const int n = bn * 128 + n_off + (j << 4) + row;
      float bs = 0.0f;
      if (bias0) bs += bias0[n];
      if (bias1) bs += bias1[n];
#pragma unroll
      for (int r = 0; r < 8; ++r)
        C[(size_t)(m0 + (mi << 4) + r) * ldc + n] = acc[mi][j][r] + bs;
    }
}

// ---------------------------------------------------------------------------
// Fused LSTM step, block-tiled: one block per 16-column hidden tile covers all
// 256 batch rows. h (256 x KS) and the 4x16 Whh gate rows staged in LDS,
// double buffered. Wave w owns batch rows [32w, 32w+32), 4 gate accs each.
// ---------------------------------------------------------------------------
__global__ void __launch_bounds__(256)
lstm_step_tiled(const __bf16* __restrict__ hprev,   // [B,H] bf16
                const __bf16* __restrict__ Whh,     // [4H,H] bf16
                const float*  __restrict__ xg,      // [B*T,4H] f32
                float*  __restrict__ c,             // [B,H] f32 (in/out)
                __bf16* __restrict__ hnext,         // [B,H] bf16
                __bf16* __restrict__ seq,           // [B*T,H] bf16 out
                int t)
{
  __shared__ __bf16 Ah[2][256][SROW];   // h rows (all batch)
  __shared__ __bf16 Bh[2][64][SROW];    // Whh rows: [gate*16 + nrow]

  const int tid  = threadIdx.x;
  const int lane = tid & 31;
  const int widx = tid >> 5;
  const int nt   = blockIdx.x;          // 0..47  (H/16)
  const int row  = lane & 15, grp = lane >> 4;

  auto stage = [&](int buf, int k0) {
#pragma unroll
    for (int i = 0; i < 4; ++i) {
      const int idx = i * 256 + tid;    // 0..1023
      const int r   = idx >> 2;         // 0..255
      const int sg  = idx & 3;
      stage16(hprev + (size_t)r * cH + k0 + sg * 8, &Ah[buf][r][sg * 8]);
    }
    {
      const int r  = tid >> 2;          // 0..63
      const int sg = tid & 3;
      const int gr = (r >> 4) * cH + (nt << 4) + (r & 15);
      stage16(Whh + (size_t)gr * cH + k0 + sg * 8, &Bh[buf][r][sg * 8]);
    }
  };

  v8f acc[2][4] = {};
  constexpr int nst = cH / KS;          // 24
  stage(0, 0);
  for (int s = 0; s < nst; ++s) {
    const int buf = s & 1;
    if (s + 1 < nst) { stage(buf ^ 1, (s + 1) * KS); WAIT_ASYNC(5); }
    else             { WAIT_ASYNC(0); }
    __syncthreads();

    Frag fb[4];
#pragma unroll
    for (int g = 0; g < 4; ++g) {
      const __bf16* p = &Bh[buf][(g << 4) + row][grp * 16];
      fb[g].h[0] = *(const u16x8*)(p);
      fb[g].h[1] = *(const u16x8*)(p + 8);
    }
#pragma unroll
    for (int mi = 0; mi < 2; ++mi) {
      v16bf fa = frag_a_lds(&Ah[buf][(widx << 5) + (mi << 4) + row][0], grp);
#pragma unroll
      for (int g = 0; g < 4; ++g)
        acc[mi][g] = __builtin_amdgcn_wmma_f32_16x16x32_bf16(
            false, fa, false, fb[g].v, (short)0, acc[mi][g], false, false);
    }
    __syncthreads();
  }

  const int n = (nt << 4) + row;
#pragma unroll
  for (int mi = 0; mi < 2; ++mi) {
#pragma unroll
    for (int r = 0; r < 8; ++r) {
      const int m = (widx << 5) + (mi << 4) + (grp << 3) + r;
      const float* xr = xg + ((size_t)m * cT + t) * cG4;
      const float iv = acc[mi][0][r] + xr[0 * cH + n];
      const float fv = acc[mi][1][r] + xr[1 * cH + n];
      const float gv = acc[mi][2][r] + xr[2 * cH + n];
      const float ov = acc[mi][3][r] + xr[3 * cH + n];
      const size_t sidx = (size_t)m * cH + n;
      const float cn = sigm(fv) * c[sidx] + sigm(iv) * tanh_fast(gv);
      const float hn = sigm(ov) * tanh_fast(cn);
      c[sidx] = cn;
      const __bf16 hb = (__bf16)hn;
      hnext[sidx] = hb;
      seq[((size_t)m * cT + t) * cH + n] = hb;
    }
  }
}

// ---------------------------------------------------------------------------
// Support kernels
// ---------------------------------------------------------------------------
__global__ void embed_bf16(const int* __restrict__ idx, const float* __restrict__ emb,
                           __bf16* __restrict__ out, long ntok, int width)
{
  long i = (long)blockIdx.x * blockDim.x + threadIdx.x;
  if (i >= ntok * width) return;
  long tok = i / width;
  int  e   = (int)(i - tok * width);
  out[i] = (__bf16)emb[(size_t)idx[tok] * width + e];
}

__global__ void f32_to_bf16(const float* __restrict__ in, __bf16* __restrict__ out, long n)
{
  long i = (long)blockIdx.x * blockDim.x + threadIdx.x;
  if (i < n) out[i] = (__bf16)in[i];
}

__global__ void zero_u32(unsigned int* __restrict__ p, long n)
{
  long i = (long)blockIdx.x * blockDim.x + threadIdx.x;
  if (i < n) p[i] = 0u;
}

// ---------------------------------------------------------------------------
// Host orchestration
// ---------------------------------------------------------------------------
extern "C" void kernel_launch(void* const* d_in, const int* in_sizes, int n_in,
                              void* d_out, int out_size, void* d_ws, size_t ws_size,
                              hipStream_t stream)
{
  (void)in_sizes; (void)n_in; (void)out_size; (void)ws_size;

  const int*   src     = (const int*)  d_in[0];
  const int*   trg     = (const int*)  d_in[1];
  const float* enc_emb = (const float*)d_in[2];
  const float* enc_Wih = (const float*)d_in[3];
  const float* enc_Whh = (const float*)d_in[4];
  const float* enc_bih = (const float*)d_in[5];
  const float* enc_bhh = (const float*)d_in[6];
  const float* dec_emb = (const float*)d_in[7];
  const float* dec_Wih = (const float*)d_in[8];
  const float* dec_Whh = (const float*)d_in[9];
  const float* dec_bih = (const float*)d_in[10];
  const float* dec_bhh = (const float*)d_in[11];
  const float* fc_W    = (const float*)d_in[12];
  const float* fc_b    = (const float*)d_in[13];
  float* out = (float*)d_out;

  // Workspace carve-out (256B aligned slices)
  char* wp = (char*)d_ws;
  auto take = [&](size_t bytes) -> void* {
    void* p = (void*)wp;
    wp += (bytes + 255) & ~(size_t)255;
    return p;
  };
  __bf16* seqA = (__bf16*)take((size_t)cMTOK * cE * 2);
  __bf16* seqB = (__bf16*)take((size_t)cMTOK * cH * 2);
  float*  xg   = (float*) take((size_t)cMTOK * cG4 * 4);
  __bf16* eWih = (__bf16*)take((size_t)cL * cG4 * cE * 2);
  __bf16* eWhh = (__bf16*)take((size_t)cL * cG4 * cH * 2);
  __bf16* dWih = (__bf16*)take((size_t)cL * cG4 * cE * 2);
  __bf16* dWhh = (__bf16*)take((size_t)cL * cG4 * cH * 2);
  __bf16* fcWb = (__bf16*)take((size_t)cV * cH * 2);
  __bf16* hbuf = (__bf16*)take((size_t)cL * 2 * cB * cH * 2);  // [layer][parity][B,H]
  float*  cbuf = (float*) take((size_t)cL * cB * cH * 4);      // [layer][B,H]

  auto cdiv = [](long a, long b) -> int { return (int)((a + b - 1) / b); };

  // Weight / FC conversion to bf16
  const long nW = (long)cL * cG4 * cE;   // == cL*cG4*cH (E==H)
  f32_to_bf16<<<cdiv(nW, 256), 256, 0, stream>>>(enc_Wih, eWih, nW);
  f32_to_bf16<<<cdiv(nW, 256), 256, 0, stream>>>(enc_Whh, eWhh, nW);
  f32_to_bf16<<<cdiv(nW, 256), 256, 0, stream>>>(dec_Wih, dWih, nW);
  f32_to_bf16<<<cdiv(nW, 256), 256, 0, stream>>>(dec_Whh, dWhh, nW);
  f32_to_bf16<<<cdiv((long)cV * cH, 256), 256, 0, stream>>>(fc_W, fcWb, (long)cV * cH);

  // Zero initial states (h bf16 ping-pong buffers + c f32)
  {
    long hw = (long)cL * 2 * cB * cH / 2;  // u32 words
    long cw = (long)cL * cB * cH;          // u32 words
    zero_u32<<<cdiv(hw, 256), 256, 0, stream>>>((unsigned int*)hbuf, hw);
    zero_u32<<<cdiv(cw, 256), 256, 0, stream>>>((unsigned int*)cbuf, cw);
  }

  // Embed source
  embed_bf16<<<cdiv(cMTOK * cE, 256), 256, 0, stream>>>(src, enc_emb, seqA, cMTOK, cE);

  int par[2] = {0, 0};  // h ping-pong parity per layer (carried encoder -> decoder)

  auto run_layer = [&](const __bf16* inseq, __bf16* outseq,
                       const __bf16* Wihb, const __bf16* Whhb,
                       const float* bih, const float* bhh, int l) {
    // xg = inseq @ Wih^T + bih + bhh   (block-tiled WMMA GEMM)
    const int gemmBlocks = (int)((cMTOK / 128) * (cG4 / 128));
    gemm_tile<<<gemmBlocks, 256, 0, stream>>>(inseq, cE, Wihb, cE, bih, bhh,
                                              xg, cG4, (int)cMTOK, cG4, cE);
    __bf16* hb0 = hbuf + ((size_t)l * 2 + 0) * cB * cH;
    __bf16* hb1 = hbuf + ((size_t)l * 2 + 1) * cB * cH;
    float*  cl  = cbuf + (size_t)l * cB * cH;
    for (int t = 0; t < cT; ++t) {
      __bf16* hp = par[l] ? hb1 : hb0;
      __bf16* hn = par[l] ? hb0 : hb1;
      lstm_step_tiled<<<cH / 16, 256, 0, stream>>>(hp, Whhb, xg, cl, hn, outseq, t);
      par[l] ^= 1;
    }
  };

  // Encoder (zero init states)
  run_layer(seqA, seqB, eWih, eWhh, enc_bih, enc_bhh, 0);
  run_layer(seqB, seqA, eWih + (size_t)cG4 * cE, eWhh + (size_t)cG4 * cH,
            enc_bih + cG4, enc_bhh + cG4, 1);

  // Decoder: embed trg, continue from encoder-final (h,c) left in hbuf/cbuf
  embed_bf16<<<cdiv(cMTOK * cE, 256), 256, 0, stream>>>(trg, dec_emb, seqA, cMTOK, cE);
  run_layer(seqA, seqB, dWih, dWhh, dec_bih, dec_bhh, 0);
  run_layer(seqB, seqA, dWih + (size_t)cG4 * cE, dWhh + (size_t)cG4 * cH,
            dec_bih + cG4, dec_bhh + cG4, 1);

  // Final FC: logits = dec_out @ fc_W^T + fc_b -> d_out [B,T,V] f32
  const int fcBlocks = (int)((cMTOK / 128) * (cV / 128));
  gemm_tile<<<fcBlocks, 256, 0, stream>>>(seqA, cH, fcWb, cH, fc_b, nullptr,
                                          out, cV, (int)cMTOK, cV, cH);
}
